// SelfAttention_73048803770788
// MI455X (gfx1250) — matrix operand order repaired and checked by reading, not compiled, not run on hardware
//
#include <hip/hip_runtime.h>
#include <hip/hip_bf16.h>

// ---------------------------------------------------------------------------
// Self-attention, B=4 S=4096 D_IN=D_H=1024, f16 WMMA compute / f32 accumulate.
// Pipeline: cast x->f16, transpose-cast W->f16, QKV GEMM (V stored transposed),
// flash-attention with online softmax. All matmuls via v_wmma_f32_16x16x32_f16.
// ---------------------------------------------------------------------------

typedef __attribute__((ext_vector_type(16))) _Float16 v16h;
typedef __attribute__((ext_vector_type(8)))  _Float16 v8h;
typedef __attribute__((ext_vector_type(4)))  _Float16 v4h;
typedef __attribute__((ext_vector_type(8)))  float    v8f;
typedef __attribute__((ext_vector_type(4)))  int      v4i;

#define BATCH 4
#define SEQ   4096
#define DIM   1024

// ---- CDNA5 async global->LDS path (guarded; falls back to load+ds_store) ---
#if defined(__gfx1250__) && __has_builtin(__builtin_amdgcn_global_load_async_to_lds_b128) && __has_builtin(__builtin_amdgcn_s_wait_asynccnt)
#define USE_ASYNC_LDS 1
#else
#define USE_ASYNC_LDS 0
#endif

__device__ __forceinline__ void gload_lds_b128(const _Float16* g, _Float16* l) {
#if USE_ASYNC_LDS
  // Builtin prototype is (v4i addrspace(1)*, v4i addrspace(3)*, imm, imm).
  // generic->AS1 / generic->AS3 via integer round-trip (LDS generic ptr low
  // 32 bits are the LDS offset on AMDGPU).
  __builtin_amdgcn_global_load_async_to_lds_b128(
      (__attribute__((address_space(1))) v4i*)(unsigned long long)(size_t)g,
      (__attribute__((address_space(3))) v4i*)(unsigned int)(size_t)l,
      0, 0);
#else
  *(uint4*)l = *(const uint4*)g;
#endif
}

__device__ __forceinline__ void wait_async_all() {
#if USE_ASYNC_LDS
  __builtin_amdgcn_s_wait_asynccnt(0);
#endif
}

// ---- WMMA wrapper: D = A(16x32 f16) * B(32x16 f16) + C(16x16 f32) ----------
__device__ __forceinline__ v8f wmma16(v16h a, v16h b, v8f c) {
  return __builtin_amdgcn_wmma_f32_16x16x32_f16(
      /*neg_a=*/false, a, /*neg_b=*/false, b,
      /*c_mod=*/(short)0, c, /*reuse_a=*/false, /*reuse_b=*/false);
}

// ===========================================================================
// Kernel 1: f32 -> f16 elementwise cast (x)
// ===========================================================================
__global__ __launch_bounds__(256) void cast_f16_kernel(
    const float* __restrict__ src, _Float16* __restrict__ dst, int n) {
  int i = (blockIdx.x * 256 + threadIdx.x) * 4;
  if (i + 3 < n) {
    float4 v = *(const float4*)(src + i);
    v4h h; h.x = (_Float16)v.x; h.y = (_Float16)v.y;
    h.z = (_Float16)v.z; h.w = (_Float16)v.w;
    *(v4h*)(dst + i) = h;
  }
}

// ===========================================================================
// Kernel 2: W[k][n] f32 -> Wt[n][k] f16 (transpose-cast, 32x32 LDS tiles)
// ===========================================================================
__global__ __launch_bounds__(256) void transpose_cast_kernel(
    const float* __restrict__ W, _Float16* __restrict__ Wt) {
  __shared__ float tile[32][33];
  int tx = threadIdx.x, ty = threadIdx.y;            // (32, 8)
  int k0 = blockIdx.y * 32, n0 = blockIdx.x * 32;
  #pragma unroll
  for (int j = 0; j < 32; j += 8)
    tile[ty + j][tx] = W[(size_t)(k0 + ty + j) * DIM + (n0 + tx)];
  __syncthreads();
  #pragma unroll
  for (int j = 0; j < 32; j += 8)
    Wt[(size_t)(n0 + ty + j) * DIM + (k0 + tx)] = (_Float16)tile[tx][ty + j];
}

// ===========================================================================
// Kernel 3: QKV projection GEMM.  Out[m,n] = sum_k X[m,k]*Wt[n,k] + bias[n]
//   X:  [16384, 1024] f16 row-major
//   Wt: [1024 n, 1024 k] f16 (pre-transposed weights)
//   out: f16; transposeOut==0 -> [m, n]; ==1 -> Vt layout [b][n][s]
// Block: 256 thr = 8 waves; tile M=128 N=128 K=32, double-buffered LDS.
// Wave (w>>1 -> 32-row strip, w&1 -> 64-col strip) = 2x4 WMMA subtiles.
// ===========================================================================
#define GT_M 128
#define GT_N 128
#define GT_K 32

__global__ __launch_bounds__(256) void qkv_gemm_kernel(
    const _Float16* __restrict__ X, const _Float16* __restrict__ Wt,
    const float* __restrict__ bias, _Float16* __restrict__ out,
    int transposeOut) {
  __shared__ _Float16 As[2][GT_M * GT_K];   // 8 KB each
  __shared__ _Float16 Bs[2][GT_N * GT_K];   // Bs[n][k]

  const int tid = threadIdx.x, lane = tid & 31, w = tid >> 5;
  const int mBase = blockIdx.y * GT_M, nBase = blockIdx.x * GT_N;
  const int wm = (w >> 1) * 32, wn = (w & 1) * 64;

  v8f acc[2][4] = {};

  // tile loader: 256 thr * b128 (8 halfs) = 2048 halfs/round; 2 rounds each
  const int lrow = tid >> 2, lcol = (tid & 3) * 8;
  auto loadTiles = [&](int buf, int kBase) {
    #pragma unroll
    for (int rr = 0; rr < 2; ++rr) {
      int row = lrow + rr * 64;
      gload_lds_b128(&X [(size_t)(mBase + row) * DIM + kBase + lcol],
                     &As[buf][row * GT_K + lcol]);
      gload_lds_b128(&Wt[(size_t)(nBase + row) * DIM + kBase + lcol],
                     &Bs[buf][row * GT_K + lcol]);
    }
  };

  loadTiles(0, 0);
  wait_async_all();
  __syncthreads();

  const int r   = lane & 15;          // A row / B col within 16
  const int cb  = (lane >> 4) * 16;   // A: K half base (0 or 16)
  const int kh  = (lane >> 4) * 16;   // B: K half base (0 or 16)
  const int NK  = DIM / GT_K;         // 32 k-steps

  for (int ks = 0; ks < NK; ++ks) {
    const int cur = ks & 1;
    if (ks + 1 < NK) loadTiles(cur ^ 1, (ks + 1) * GT_K);

    v16h afr[2], bfr[4];
    #pragma unroll
    for (int mi = 0; mi < 2; ++mi) {
      const _Float16* ap = &As[cur][(wm + mi * 16 + r) * GT_K];
      union { v16h v; v8h h[2]; } u;
      u.h[0] = *(const v8h*)(ap + cb);        // K = cb..cb+7
      u.h[1] = *(const v8h*)(ap + cb + 8);    // K = cb+8..cb+15
      afr[mi] = u.v;
    }
    #pragma unroll
    for (int ni = 0; ni < 4; ++ni)
      bfr[ni] = *(const v16h*)(&Bs[cur][(wn + ni * 16 + r) * GT_K + kh]);

    #pragma unroll
    for (int mi = 0; mi < 2; ++mi)
      #pragma unroll
      for (int ni = 0; ni < 4; ++ni)
        acc[mi][ni] = wmma16(afr[mi], bfr[ni], acc[mi][ni]);

    wait_async_all();
    __syncthreads();
  }

  // Epilogue: C layout -> element (m = vgpr + (lane>>4)*8, n = lane&15)
  const int r8 = (lane >> 4) * 8, nlo = lane & 15;
  #pragma unroll
  for (int mi = 0; mi < 2; ++mi)
    #pragma unroll
    for (int ni = 0; ni < 4; ++ni) {
      const int n = nBase + wn + ni * 16 + nlo;
      const float bv = bias[n];
      #pragma unroll
      for (int rr = 0; rr < 8; ++rr) {
        const int m = mBase + wm + mi * 16 + r8 + rr;
        const _Float16 h = (_Float16)(acc[mi][ni][rr] + bv);
        if (!transposeOut) {
          out[(size_t)m * DIM + n] = h;                       // [B*S, D]
        } else {
          const int b = m >> 12, s = m & (SEQ - 1);
          out[(((size_t)b * DIM + n) << 12) + s] = h;         // Vt [b][n][s]
        }
      }
    }
}

// ===========================================================================
// Kernel 4: flash attention.
//  Block = 256 thr (8 waves), handles (batch b, 16 query rows).
//  Per 64-key tile:
//    waves: ntile = w&3 (16-key subtile), khalf = w>>2 (512-wide k half)
//    -> partial 16x16 scores to LDS (2 partial buffers summed in softmax)
//    online softmax in LDS; P (f16) to LDS
//    PV: wave w owns h-chunk [w*128, w*128+128); O accum = 8 x v8f regs.
//  K, Vt fragments read straight from global (Q/K/V f16 = 96 MB, L2-resident).
// ===========================================================================
__global__ __launch_bounds__(256) void attention_kernel(
    const _Float16* __restrict__ Qh,   // [B*S, D]
    const _Float16* __restrict__ Kh,   // [B*S, D]
    const _Float16* __restrict__ Vt,   // [B, D, S]
    float* __restrict__ out) {         // [B, S, D] f32
  __shared__ _Float16 Qs[16 * DIM];        // 32 KB
  __shared__ float    sc[2][16 * 64];      // 8 KB partial scores
  __shared__ _Float16 Ps[16 * 64];         // 2 KB probabilities
  __shared__ float    m_run[16], l_run[16], corr[16], red[16 * 16];

  const int tid = threadIdx.x, lane = tid & 31, w = tid >> 5;
  const int q0 = blockIdx.x * 16;
  const int b  = blockIdx.y;
  const float scale = 0.03125f;   // 1/sqrt(1024)

  // Load 16 x 1024 Q tile (32 KB): 8 rounds of 256 x b128
  #pragma unroll
  for (int rr = 0; rr < 8; ++rr) {
    const int off = rr * 2048 + tid * 8;
    const int row = off >> 10, col = off & (DIM - 1);
    gload_lds_b128(&Qh[(size_t)(b * SEQ + q0 + row) * DIM + col], &Qs[off]);
  }
  if (tid < 16) { m_run[tid] = -3.0e38f; l_run[tid] = 0.0f; }
  wait_async_all();
  __syncthreads();

  v8f o[8] = {};
  const int ntile = w & 3, khalf = w >> 2;
  const int r    = lane & 15;
  const int cb   = (lane >> 4) * 16;
  const int kh16 = (lane >> 4) * 16;
  const int hb   = w * 128;
  const int sr   = tid & 15, sg = tid >> 4;   // softmax row / col-group

  for (int kt = 0; kt < SEQ / 64; ++kt) {
    const int key0 = kt * 64;

    // ---- QK^T partial: 16 q x 16 keys over a 512-wide k half -------------
    v8f s8 = {};
    const _Float16* kbase =
        &Kh[(size_t)(b * SEQ + key0 + ntile * 16 + r) * DIM + kh16];
    #pragma unroll 4
    for (int ks = 0; ks < 16; ++ks) {
      const int kk = khalf * 512 + ks * 32;
      union { v16h v; v8h h[2]; } ua;
      const _Float16* ap = &Qs[r * DIM + kk];
      ua.h[0] = *(const v8h*)(ap + cb);
      ua.h[1] = *(const v8h*)(ap + cb + 8);
      const v16h bv = *(const v16h*)(kbase + kk);
      s8 = wmma16(ua.v, bv, s8);
    }
    #pragma unroll
    for (int rr = 0; rr < 8; ++rr)
      sc[khalf][((lane >> 4) * 8 + rr) * 64 + ntile * 16 + r] = s8[rr];
    __syncthreads();

    // ---- online softmax over this 16x64 tile ----------------------------
    float v0[4], lm = -3.0e38f;
    #pragma unroll
    for (int j = 0; j < 4; ++j) {
      const int idx = sr * 64 + sg * 4 + j;
      v0[j] = (sc[0][idx] + sc[1][idx]) * scale;
      lm = fmaxf(lm, v0[j]);
    }
    red[sr * 16 + sg] = lm;
    __syncthreads();
    if (tid < 16) {
      float mt = red[tid * 16];
      #pragma unroll
      for (int j = 1; j < 16; ++j) mt = fmaxf(mt, red[tid * 16 + j]);
      const float mo = m_run[tid];
      const float mn = fmaxf(mo, mt);
      const float c  = __expf(mo - mn);
      corr[tid] = c; l_run[tid] *= c; m_run[tid] = mn;
    }
    __syncthreads();
    const float mrow = m_run[sr];
    float psum = 0.0f;
    #pragma unroll
    for (int j = 0; j < 4; ++j) {
      const float p = __expf(v0[j] - mrow);
      Ps[sr * 64 + sg * 4 + j] = (_Float16)p;
      psum += p;
    }
    red[sr * 16 + sg] = psum;
    __syncthreads();
    if (tid < 16) {
      float st = 0.0f;
      #pragma unroll
      for (int j = 0; j < 16; ++j) st += red[tid * 16 + j];
      l_run[tid] += st;
    }

    // ---- PV: rescale O by corr, accumulate P(16x64) * V(64 x 128-chunk) --
    float cf[8];
    #pragma unroll
    for (int rr = 0; rr < 8; ++rr) cf[rr] = corr[(lane >> 4) * 8 + rr];
    #pragma unroll
    for (int ni = 0; ni < 8; ++ni)
      #pragma unroll
      for (int rr = 0; rr < 8; ++rr) o[ni][rr] *= cf[rr];

    #pragma unroll
    for (int kk = 0; kk < 64; kk += 32) {
      union { v16h v; v8h h[2]; } ua;
      const _Float16* pp = &Ps[r * 64 + kk];
      ua.h[0] = *(const v8h*)(pp + cb);
      ua.h[1] = *(const v8h*)(pp + cb + 8);
      #pragma unroll
      for (int ni = 0; ni < 8; ++ni) {
        const _Float16* vb =
            &Vt[((size_t)b * DIM + hb + ni * 16 + r) * SEQ + key0 + kk + kh16];
        const v16h bv = *(const v16h*)vb;
        o[ni] = wmma16(ua.v, bv, o[ni]);
      }
    }
  }

  __syncthreads();
  float linv[8];
  #pragma unroll
  for (int rr = 0; rr < 8; ++rr)
    linv[rr] = 1.0f / l_run[(lane >> 4) * 8 + rr];
  #pragma unroll
  for (int ni = 0; ni < 8; ++ni) {
    const int h = hb + ni * 16 + r;
    #pragma unroll
    for (int rr = 0; rr < 8; ++rr) {
      const int m = (lane >> 4) * 8 + rr;
      out[(size_t)(b * SEQ + q0 + m) * DIM + h] = o[ni][rr] * linv[rr];
    }
  }
}

// ===========================================================================
// Launch
// ===========================================================================
extern "C" void kernel_launch(void* const* d_in, const int* in_sizes, int n_in,
                              void* d_out, int out_size, void* d_ws, size_t ws_size,
                              hipStream_t stream) {
  const float* x  = (const float*)d_in[0];
  const float* Wq = (const float*)d_in[1];
  const float* bq = (const float*)d_in[2];
  const float* Wk = (const float*)d_in[3];
  const float* bk = (const float*)d_in[4];
  const float* Wv = (const float*)d_in[5];
  const float* bv = (const float*)d_in[6];
  float* out = (float*)d_out;

  const size_t M   = (size_t)BATCH * SEQ;      // 16384 tokens
  const size_t nX  = M * DIM;                  // 16M elems
  const size_t nW  = (size_t)DIM * DIM;        // 1M elems

  // workspace layout (f16): xh | WqT | WkT | WvT | Qh | Kh | Vt  (~134 MB)
  _Float16* p   = (_Float16*)d_ws;
  _Float16* xh  = p; p += nX;
  _Float16* WqT = p; p += nW;
  _Float16* WkT = p; p += nW;
  _Float16* WvT = p; p += nW;
  _Float16* Qh  = p; p += nX;
  _Float16* Kh  = p; p += nX;
  _Float16* Vt  = p; p += nX;
  (void)ws_size; (void)in_sizes; (void)n_in; (void)out_size;

  // 1) cast x -> f16
  cast_f16_kernel<<<dim3(nX / (256 * 4)), dim3(256), 0, stream>>>(x, xh, (int)nX);

  // 2) transpose-cast weights -> Wt[n][k] f16
  dim3 tg(DIM / 32, DIM / 32), tb(32, 8);
  transpose_cast_kernel<<<tg, tb, 0, stream>>>(Wq, WqT);
  transpose_cast_kernel<<<tg, tb, 0, stream>>>(Wk, WkT);
  transpose_cast_kernel<<<tg, tb, 0, stream>>>(Wv, WvT);

  // 3) QKV projections (V stored transposed for PV B-fragments)
  dim3 gg(DIM / GT_N, M / GT_M), gb(256);
  qkv_gemm_kernel<<<gg, gb, 0, stream>>>(xh, WqT, bq, Qh, 0);
  qkv_gemm_kernel<<<gg, gb, 0, stream>>>(xh, WkT, bk, Kh, 0);
  qkv_gemm_kernel<<<gg, gb, 0, stream>>>(xh, WvT, bv, Vt, 1);

  // 4) flash attention
  attention_kernel<<<dim3(SEQ / 16, BATCH), dim3(256), 0, stream>>>(Qh, Kh, Vt, out);
}
